// ModulatedConv2D_52226802319456
// MI455X (gfx1250) — matrix operand order repaired and checked by reading, not compiled, not run on hardware
//
#include <hip/hip_runtime.h>

// ---------------------------------------------------------------------------
// ModulatedConv2D (StyleGAN2-style) for MI455X / gfx1250.
// Reference == one dense conv: rows [256, 64, 256] (rows=B*H), filter
// [12,3,256,256], SAME padding (pad_top=5, pad_left=1).
// Implicit GEMM M=16384, N=256, K=9216; bf16 WMMA, f32 accumulate.
// Weights are DMA'd into LDS with the Tensor Data Mover (double buffered,
// TENSORcnt synchronized); activations staged per row in LDS as bf16.
// ---------------------------------------------------------------------------

typedef __attribute__((ext_vector_type(16))) __bf16        v16bf;
typedef __attribute__((ext_vector_type(8)))  float         v8f;
typedef __attribute__((ext_vector_type(4)))  unsigned int  u32x4;
typedef __attribute__((ext_vector_type(2)))  unsigned int  u32x2;
typedef __attribute__((ext_vector_type(4)))  float         f32x4;
typedef __attribute__((ext_vector_type(8)))  int           i32x8;
typedef __attribute__((ext_vector_type(4)))  int           i32x4;

union ABu { v16bf v; u32x4 q[2]; };

#define XROW_BYTES 33792              // 66 cols x 256 ch x bf16
#define BBUF_ELEMS 32768              // 256 f x 128 c bf16
#define BBUF_BYTES 65536
#define SMEM_BYTES (XROW_BYTES + 2 * BBUF_BYTES)   // 164864 < 320 KB/WGP

#if defined(__HIP_DEVICE_COMPILE__) && __has_builtin(__builtin_amdgcn_tensor_load_to_lds)
#define USE_TDM 1
#else
#define USE_TDM 0
#endif

__device__ __forceinline__ unsigned short f2bf(float f) {
  unsigned int u = __builtin_bit_cast(unsigned int, f);
  u += 0x7FFFu + ((u >> 16) & 1u);    // round-to-nearest-even
  return (unsigned short)(u >> 16);
}

#if USE_TDM
// Issue one TDM 2D tile load: 256 rows (f) x 128 cols (c) of bf16, row stride
// 256 elements, from `gbase` into LDS byte offset `ldsOff`.  D# layout per
// CDNA5 ISA.
__device__ __forceinline__ void tdm_load_b(const unsigned short* gbase,
                                           unsigned ldsOff) {
  const unsigned long long ga = (unsigned long long)gbase;
  u32x4 g0;
  g0[0] = 1u;                                            // count = 1 valid D#
  g0[1] = ldsOff;                                        // lds_addr (bytes)
  g0[2] = (unsigned)(ga & 0xFFFFFFFFu);                  // global_addr[31:0]
  g0[3] = (unsigned)((ga >> 32) & 0x1FFFFFFu)            // global_addr[56:32]
        | (2u << 30);                                    // type = 2 ("image")
  i32x8 g1;
  g1[0] = 0x00010000;                                    // data_size = 1 (2B)
  g1[1] = (int)(256u << 16);                             // tensor_dim0 = 256
  g1[2] = (int)(256u << 16);                             // tensor_dim1 = 256
  g1[3] = (int)(128u << 16);                             // tile_dim0   = 128
  g1[4] = 256;                                           // tile_dim1   = 256
  g1[5] = 256;                                           // dim0_stride = 256
  g1[6] = 0;
  g1[7] = 0;
  i32x4 z4 = {0, 0, 0, 0};
#if __clang_major__ >= 23
  i32x8 z8 = {0, 0, 0, 0, 0, 0, 0, 0};
  __builtin_amdgcn_tensor_load_to_lds(g0, g1, z4, z4, z8, 0);
#else
  __builtin_amdgcn_tensor_load_to_lds(g0, g1, z4, z4, 0);
#endif
}
#endif

// ---------------------------------------------------------------------------
// Prep 1: demod[b,f] = rsqrt( sum_{kh,kw,c} (kernel[kh,kw,c,f]*style[b,c])^2 + 1e-8 )
// ---------------------------------------------------------------------------
__global__ __launch_bounds__(256) void prep_demod(const float* __restrict__ kern,
                                                  const float* __restrict__ style,
                                                  float* __restrict__ demod) {
  const int id = blockIdx.x * 256 + threadIdx.x;   // 0..1023
  const int b = id >> 8;
  const int f = id & 255;
  const float* sp = style + b * 256;
  float sum = 0.f;
  for (int t = 0; t < 9; ++t) {
    const float* kp = kern + (t * 256) * 256 + f;
    #pragma unroll 4
    for (int c = 0; c < 256; ++c) {
      float v = kp[c * 256] * sp[c];
      sum += v * v;
    }
  }
  demod[id] = rsqrtf(sum + 1e-8f);
}

// ---------------------------------------------------------------------------
// Prep 2: wB[tap][f][c] = bf16( kernel[kh,fw,c,f] * style[b,c] * demod[b,f] )
// tap = (b*3+kh)*3+fw.  f-major / c-contiguous (pre-transposed for WMMA B).
// ---------------------------------------------------------------------------
__global__ __launch_bounds__(256) void prep_weights(const float* __restrict__ kern,
                                                    const float* __restrict__ style,
                                                    const float* __restrict__ demod,
                                                    unsigned short* __restrict__ wB) {
  const int g   = blockIdx.x * 256 + threadIdx.x;  // == tap*65536 + f*256 + c
  const int c   = g & 255;
  const int f   = (g >> 8) & 255;
  const int tap = g >> 16;                          // 0..35
  const int b   = tap / 9;
  const int kh  = (tap / 3) % 3;
  const int fw  = tap % 3;
  const float v = kern[((kh * 3 + fw) * 256 + c) * 256 + f]
                * style[b * 256 + c] * demod[b * 256 + f];
  wB[g] = f2bf(v);
}

// ---------------------------------------------------------------------------
// Main: 256 blocks (one merged output row i each), 256 threads = 8 waves.
// Chunk = (fh, fw, half): B sub-tensor 256 f x 128 c staged in LDS by TDM,
// double buffered.  Per chunk each wave runs 4 K-steps x 8 N-tiles = 32 WMMAs.
// ---------------------------------------------------------------------------
__global__ __launch_bounds__(256) void conv_wmma(const float* __restrict__ x,
                                                 const unsigned short* __restrict__ wB,
                                                 float* __restrict__ out) {
  extern __shared__ __align__(16) unsigned short smem[];
  unsigned short* xrow = smem;                       // [66][256] bf16 (+halo)
  unsigned short* bB   = smem + XROW_BYTES / 2;      // 2 x [256 f][128 c] bf16

  const int i    = blockIdx.x;                       // merged row = b*64+h
  const int tid  = threadIdx.x;
  const int lane = tid & 31;
  const int wave = tid >> 5;
  const int mW   = (wave & 3) * 16;                  // w base of M-tile
  const int nB   = (wave >> 2) * 128;                // f base of N range
  const int lm   = lane & 15;
  const int lh   = lane >> 4;

  // zero halo columns 0 and 65 once
  if (tid < 128) {
    ((unsigned int*)xrow)[tid] = 0u;
    ((unsigned int*)(xrow + 65 * 256))[tid] = 0u;
  }

  v8f acc[8];
  #pragma unroll
  for (int nt = 0; nt < 8; ++nt)
    #pragma unroll
    for (int e = 0; e < 8; ++e) acc[nt][e] = 0.f;

  // valid fh's form a contiguous range (r = i + fh - 5 in [0,255])
  const int fhLo = (i < 5) ? (5 - i) : 0;
  const int fhHi = (i > 249) ? (260 - i) : 11;
  const int nCh  = (fhHi - fhLo + 1) * 6;            // 6 chunks per fh

#if USE_TDM
  if (wave == 0) {                                   // prologue: chunk 0
    tdm_load_b(wB + ((fhLo * 3 + 0) << 16) + 0, XROW_BYTES);
  }
#endif

  for (int ch = 0; ch < nCh; ++ch) {
    const int rel  = ch % 6;
    const int fh   = fhLo + ch / 6;
    const int fw   = rel >> 1;
    const int half = rel & 1;

    if (rel == 0) {
      // stage x row r (fp32 -> bf16); prior compute done (trailing barrier)
      const int r = i + fh - 5;
      const f32x4* src = (const f32x4*)(x + r * (64 * 256) + tid * 64);
      unsigned short* dst = xrow + 256 + tid * 64;   // +256: left halo col
      #pragma unroll
      for (int j = 0; j < 16; ++j) {
        f32x4 v = src[j];
        u32x2 p;
        p.x = (unsigned int)f2bf(v.x) | ((unsigned int)f2bf(v.y) << 16);
        p.y = (unsigned int)f2bf(v.z) | ((unsigned int)f2bf(v.w) << 16);
        *(u32x2*)(dst + j * 4) = p;
      }
    }

#if USE_TDM
    if (wave == 0) {
      if (ch + 1 < nCh) {                            // prefetch next chunk
        const int rel2 = (ch + 1) % 6;
        const int fh2  = fhLo + (ch + 1) / 6;
        tdm_load_b(wB + ((fh2 * 3 + (rel2 >> 1)) << 16) + (rel2 & 1) * 128,
                   XROW_BYTES + ((ch + 1) & 1) * BBUF_BYTES);
        __builtin_amdgcn_s_wait_tensorcnt(1);        // current chunk landed
      } else {
        __builtin_amdgcn_s_wait_tensorcnt(0);
      }
    }
    const unsigned short* bbase = bB + (ch & 1) * BBUF_ELEMS;
#else
    {
      // fallback: cooperative global->LDS copy of current chunk
      const unsigned short* src =
          wB + ((fh * 3 + fw) << 16) + half * 128 + tid * 256;
      unsigned short* dst = bB + tid * 128;
      #pragma unroll
      for (int j = 0; j < 8; ++j)
        *(u32x4*)(dst + j * 8) = *(const u32x4*)(src + j * 8);
    }
    const unsigned short* bbase = bB;
#endif
    __syncthreads();                                 // xrow + B chunk visible

    #pragma unroll
    for (int kcl = 0; kcl < 4; ++kcl) {
      const int c0 = half * 128 + kcl * 32;

      // A-tile (16x32 bf16) from LDS, ISA A layout:
      // lane lm: K {0..7,16..23}; lane lm+16: K {8..15,24..31}
      ABu a;
      const unsigned short* ap = xrow + (mW + lm + fw) * 256 + c0 + lh * 8;
      a.q[0] = *(const u32x4*)ap;
      a.q[1] = *(const u32x4*)(ap + 16);

      #pragma unroll
      for (int nt = 0; nt < 8; ++nt) {
        // B-tile (32x16 bf16) from LDS chunk [f][128]: lane = column f,
        // lanes 0-15 hold K {0..7,16..23}, lanes 16-31 hold K {8..15,24..31}
        ABu bt;
        const unsigned short* bp =
            bbase + (nB + nt * 16 + lm) * 128 + kcl * 32 + lh * 8;
        bt.q[0] = *(const u32x4*)bp;
        bt.q[1] = *(const u32x4*)(bp + 16);

        acc[nt] = __builtin_amdgcn_wmma_f32_16x16x32_bf16(
            false, a.v, false, bt.v, (short)0, acc[nt], false, false);
      }
    }
    __syncthreads();                                 // buffer may be reused
  }

  // Store: C/D layout -> VGPR v: lanes 0-15 = row v, lanes 16-31 = row v+8
  float* o = out + (i * 64 + mW) * 256 + nB;
  #pragma unroll
  for (int nt = 0; nt < 8; ++nt)
    #pragma unroll
    for (int v = 0; v < 8; ++v)
      o[(v + lh * 8) * 256 + nt * 16 + lm] = acc[nt][v];
}

// ---------------------------------------------------------------------------
extern "C" void kernel_launch(void* const* d_in, const int* in_sizes, int n_in,
                              void* d_out, int out_size, void* d_ws, size_t ws_size,
                              hipStream_t stream) {
  (void)in_sizes; (void)n_in; (void)out_size; (void)ws_size;
  const float* x     = (const float*)d_in[0];   // [4,64,64,256]
  const float* style = (const float*)d_in[1];   // [4,256]
  const float* kern  = (const float*)d_in[2];   // [3,3,256,256]
  float* out = (float*)d_out;                   // [4,64,64,256]

  float*          demod = (float*)d_ws;                           // 4 KB
  unsigned short* wBp   = (unsigned short*)((char*)d_ws + 4096);  // 4.72 MB

  prep_demod  <<<4,    256, 0, stream>>>(kern, style, demod);
  prep_weights<<<9216, 256, 0, stream>>>(kern, style, demod, wBp);
  conv_wmma   <<<256,  256, SMEM_BYTES, stream>>>(x, wBp, out);
}
